// Attention_63797444214824
// MI455X (gfx1250) — matrix operand order repaired and checked by reading, not compiled, not run on hardware
//
#include <hip/hip_runtime.h>
#include <hip/hip_bf16.h>
#include <math.h>

typedef _Float16 f16;
typedef _Float16 v16h __attribute__((ext_vector_type(16)));
typedef _Float16 v8h  __attribute__((ext_vector_type(8)));
typedef float    v8f  __attribute__((ext_vector_type(8)));

union Frag16 { v16h v; v8h h8[2]; };

#define WMMA(a, b, c) \
  __builtin_amdgcn_wmma_f32_16x16x32_f16(false, (a), false, (b), (short)0, (c), false, false)

// WGP-scope prefetch (scope field omitted == SCOPE 0 == pull into all cache
// levels, incl. WGP$). __builtin_prefetch can only reach SE/SYS scopes which
// stop at GL2 -- useless for data that is already L2-resident.
__device__ __forceinline__ void prefetch_wgp(const void* p) {
  asm volatile("global_prefetch_b8 %0, off" :: "v"(p));
}

static constexpr int Bb = 2, Ss = 2048, Ee = 1024, Hh = 16, Dd = 64;
static constexpr int Mrows = Bb * Ss;                 // 4096 tokens
static constexpr float ATTN_SCALE = 0.03125f;         // 1/sqrt(1024)
static constexpr float NEG_BIG = -1.0e30f;

// ---------------------------------------------------------------------------
// Kernel 1: f32 -> f16 staging of x and the four weight matrices
// ---------------------------------------------------------------------------
__global__ __launch_bounds__(256) void cvt_kernel(
    const float* __restrict__ x, const float* __restrict__ wq,
    const float* __restrict__ wk, const float* __restrict__ wv,
    const float* __restrict__ wo,
    f16* __restrict__ xh, f16* __restrict__ wqh, f16* __restrict__ wkh,
    f16* __restrict__ wvh, f16* __restrict__ woh) {
  size_t i = (size_t)blockIdx.x * blockDim.x + threadIdx.x;
  const size_t NX = (size_t)Mrows * Ee;   // 4,194,304
  const size_t NW = (size_t)Ee * Ee;      // 1,048,576
  if (i < NX) { xh[i] = (f16)x[i]; return; }
  size_t j = i - NX;
  if (j >= 4 * NW) return;
  size_t w = j / NW, o = j % NW;
  if (w == 0)      wqh[o] = (f16)wq[o];
  else if (w == 1) wkh[o] = (f16)wk[o];
  else if (w == 2) wvh[o] = (f16)wv[o];
  else             woh[o] = (f16)wo[o];
}

// ---------------------------------------------------------------------------
// Kernel 2: Q/K/V projections (y = x @ W^T), RoPE on Q,K.
// One wave computes a 32x16 output tile for all three matrices
// (two A fragments reuse each weight B fragment; A reused across q/k/v).
// Outputs: Qh,Kh as [B,H,S,D] f16 ; V transposed as [B,H,D,S] f16.
// ---------------------------------------------------------------------------
__global__ __launch_bounds__(256) void qkv_kernel(
    const f16* __restrict__ xh, const f16* __restrict__ wqh,
    const f16* __restrict__ wkh, const f16* __restrict__ wvh,
    f16* __restrict__ qh, f16* __restrict__ kh, f16* __restrict__ vt) {
  int w    = blockIdx.x * 8 + (threadIdx.x >> 5);
  int lane = threadIdx.x & 31;
  int col  = lane & 15;
  int half = lane >> 4;
  int mtile = w >> 6;          // 0..127  (32-row tiles)
  int ntile = w & 63;          // 0..63
  int m0 = mtile << 5, n0 = ntile << 4;

  v8f aq0 = {}, ak0 = {}, av0 = {};
  v8f aq1 = {}, ak1 = {}, av1 = {};
  const v8h* xrow0 = (const v8h*)(xh + (size_t)(m0 + col) * Ee);
  const v8h* xrow1 = (const v8h*)(xh + (size_t)(m0 + 16 + col) * Ee);
  const f16* wqrow = wqh + (size_t)(n0 + col) * Ee + 16 * half;
  const f16* wkrow = wkh + (size_t)(n0 + col) * Ee + 16 * half;
  const f16* wvrow = wvh + (size_t)(n0 + col) * Ee + 16 * half;

  for (int k0 = 0; k0 < Ee; k0 += 32) {
    prefetch_wgp(wqrow + k0 + 64);   // next-next k tile into WGP$
    prefetch_wgp(wkrow + k0 + 64);
    prefetch_wgp(wvrow + k0 + 64);
    Frag16 a0, a1;
    a0.h8[0] = xrow0[(k0 >> 3) + 2 * half];
    a0.h8[1] = xrow0[(k0 >> 3) + 2 * half + 1];
    a1.h8[0] = xrow1[(k0 >> 3) + 2 * half];
    a1.h8[1] = xrow1[(k0 >> 3) + 2 * half + 1];
    v16h bq = *(const v16h*)(wqrow + k0);
    v16h bk = *(const v16h*)(wkrow + k0);
    v16h bv = *(const v16h*)(wvrow + k0);
    aq0 = WMMA(a0.v, bq, aq0);  aq1 = WMMA(a1.v, bq, aq1);
    ak0 = WMMA(a0.v, bk, ak0);  ak1 = WMMA(a1.v, bk, ak1);
    av0 = WMMA(a0.v, bv, av0);  av1 = WMMA(a1.v, bv, av1);
  }

  // RoPE (interleaved pairs) + scatter stores.
  int ng  = n0 + col;            // global output column
  int h   = ng >> 6;
  int d   = ng & 63;
  int ip  = d >> 1;
  int par = d & 1;
  // theta_i = 10000^(-i/32) = exp(-i * ln(10000)/32)
  float theta = __expf(-(float)ip * 0.28782313662425572f);

  for (int sub = 0; sub < 2; ++sub) {
    v8f aq = sub ? aq1 : aq0;
    v8f ak = sub ? ak1 : ak0;
    v8f av = sub ? av1 : av0;
    for (int r = 0; r < 8; ++r) {
      int m = m0 + sub * 16 + r + 8 * half;   // global token row
      int b = m >> 11;
      int s = m & (Ss - 1);
      float ang = (float)s * theta;
      float cth = cosf(ang), sth = sinf(ang);

      float qv = aq[r], kv = ak[r];
      float qp = __shfl_xor(qv, 1, 32);   // pair partner (d^1)
      float kp = __shfl_xor(kv, 1, 32);
      float qo = par ? (qp * sth + qv * cth) : (qv * cth - qp * sth);
      float ko = par ? (kp * sth + kv * cth) : (kv * cth - kp * sth);

      size_t bh = (size_t)(b * Hh + h);
      qh[(bh * Ss + s) * Dd + d] = (f16)qo;
      kh[(bh * Ss + s) * Dd + d] = (f16)ko;
      vt[(bh * Dd + d) * Ss + s] = (f16)av[r];
    }
  }
}

// ---------------------------------------------------------------------------
// Kernel 3: causal flash attention. One wave per (bh, 16-row q tile),
// 32 keys per inner step: 4 WMMAs QK^T + 4 WMMAs P.V. Next K/V tiles
// are prefetched into the WGP cache while softmax runs.
// ---------------------------------------------------------------------------
__global__ __launch_bounds__(128) void attn_kernel(
    const f16* __restrict__ qh, const f16* __restrict__ kh,
    const f16* __restrict__ vt, f16* __restrict__ ah) {
  __shared__ f16 plds[4][16][32];
  int wv   = threadIdx.x >> 5;
  int w    = blockIdx.x * 4 + wv;
  int lane = threadIdx.x & 31;
  int col  = lane & 15;
  int half = lane >> 4;
  int qt = w & 127;        // q tile within (b,h)
  int bh = w >> 7;         // 0..31
  int q0 = qt << 4;

  // Q as WMMA A fragments (two K=32 chunks over D=64), loaded once.
  const v8h* qrow = (const v8h*)(qh + ((size_t)bh * Ss + q0 + col) * Dd);
  Frag16 a0, a1;
  a0.h8[0] = qrow[2 * half];     a0.h8[1] = qrow[2 * half + 1];
  a1.h8[0] = qrow[4 + 2 * half]; a1.h8[1] = qrow[5 + 2 * half];

  float mrow[8], lrow[8];
  v8f acc[4] = {};
  for (int r = 0; r < 8; ++r) { mrow[r] = NEG_BIG; lrow[r] = 0.f; }

  int kend = q0 + 15;
  for (int k0 = 0; k0 <= kend; k0 += 32) {
    // ---- scores: 16 q-rows x 32 keys
    v8f sc[2];
    for (int c = 0; c < 2; ++c) {
      const f16* kb = kh + ((size_t)bh * Ss + k0 + c * 16 + col) * Dd + 16 * half;
      v16h b0 = *(const v16h*)kb;          // d = 0..31 chunk
      v16h b1 = *(const v16h*)(kb + 32);   // d = 32..63 chunk
      v8f z = {};
      z = WMMA(a0.v, b0, z);
      z = WMMA(a1.v, b1, z);
      sc[c] = z;
    }
    // prefetch next key/value tiles (into WGP$) while we do softmax
    if (k0 + 32 <= kend) {
      prefetch_wgp(kh + ((size_t)bh * Ss + k0 + 32 + col) * Dd);
      prefetch_wgp(kh + ((size_t)bh * Ss + k0 + 48 + col) * Dd);
      prefetch_wgp(vt + ((size_t)bh * Dd + col) * Ss + k0 + 32);
      prefetch_wgp(vt + ((size_t)bh * Dd + 16 + col) * Ss + k0 + 32);
      prefetch_wgp(vt + ((size_t)bh * Dd + 32 + col) * Ss + k0 + 32);
      prefetch_wgp(vt + ((size_t)bh * Dd + 48 + col) * Ss + k0 + 32);
    }
    // ---- scale + causal mask + online softmax
    for (int r = 0; r < 8; ++r) {
      int mg = q0 + r + 8 * half;
      float s0 = sc[0][r] * ATTN_SCALE; if (k0 + col > mg)      s0 = NEG_BIG;
      float s1 = sc[1][r] * ATTN_SCALE; if (k0 + 16 + col > mg) s1 = NEG_BIG;
      float t = fmaxf(s0, s1);
      t = fmaxf(t, __shfl_xor(t, 1, 32));
      t = fmaxf(t, __shfl_xor(t, 2, 32));
      t = fmaxf(t, __shfl_xor(t, 4, 32));
      t = fmaxf(t, __shfl_xor(t, 8, 32));
      float mn = fmaxf(mrow[r], t);
      float alpha = __expf(mrow[r] - mn);
      float p0 = __expf(s0 - mn);
      float p1 = __expf(s1 - mn);
      float rs = p0 + p1;
      rs += __shfl_xor(rs, 1, 32);
      rs += __shfl_xor(rs, 2, 32);
      rs += __shfl_xor(rs, 4, 32);
      rs += __shfl_xor(rs, 8, 32);
      lrow[r] = lrow[r] * alpha + rs;
      mrow[r] = mn;
      acc[0][r] *= alpha; acc[1][r] *= alpha;
      acc[2][r] *= alpha; acc[3][r] *= alpha;
      plds[wv][r + 8 * half][col]      = (f16)p0;   // C-layout -> LDS
      plds[wv][r + 8 * half][16 + col] = (f16)p1;
    }
    __builtin_amdgcn_wave_barrier();
    // ---- reload P in A-fragment layout (same wave, DS ops are in-order)
    Frag16 ap;
    const v8h* prow = (const v8h*)&plds[wv][col][0];
    ap.h8[0] = prow[2 * half];
    ap.h8[1] = prow[2 * half + 1];
    __builtin_amdgcn_wave_barrier();
    // ---- P.V accumulate over 4 d tiles (V stored [bh, d, s])
    for (int f = 0; f < 4; ++f) {
      v16h bv = *(const v16h*)(vt + ((size_t)bh * Dd + f * 16 + col) * Ss +
                               k0 + 16 * half);
      acc[f] = WMMA(ap.v, bv, acc[f]);
    }
  }

  int b = bh >> 4, h = bh & 15;
  for (int r = 0; r < 8; ++r) {
    float inv = 1.0f / lrow[r];
    int s = q0 + r + 8 * half;
    for (int f = 0; f < 4; ++f) {
      ah[((size_t)b * Ss + s) * Ee + h * Dd + f * 16 + col] =
          (f16)(acc[f][r] * inv);
    }
  }
}

// ---------------------------------------------------------------------------
// Kernel 4: out = attn @ Wo^T, f32 output. One wave per 32x64 tile
// (two A fragments reuse each Wo B fragment).
// ---------------------------------------------------------------------------
__global__ __launch_bounds__(256) void proj_kernel(
    const f16* __restrict__ ah, const f16* __restrict__ woh,
    float* __restrict__ out) {
  int w    = blockIdx.x * 8 + (threadIdx.x >> 5);
  int lane = threadIdx.x & 31;
  int col  = lane & 15;
  int half = lane >> 4;
  int mtile = w >> 4;        // 0..127  (32-row tiles)
  int ng    = w & 15;        // 64-col group
  int m0 = mtile << 5, n0 = ng << 6;

  v8f acc0[4] = {}, acc1[4] = {};
  const v8h* arow0 = (const v8h*)(ah + (size_t)(m0 + col) * Ee);
  const v8h* arow1 = (const v8h*)(ah + (size_t)(m0 + 16 + col) * Ee);
  for (int k0 = 0; k0 < Ee; k0 += 32) {
    Frag16 a0, a1;
    a0.h8[0] = arow0[(k0 >> 3) + 2 * half];
    a0.h8[1] = arow0[(k0 >> 3) + 2 * half + 1];
    a1.h8[0] = arow1[(k0 >> 3) + 2 * half];
    a1.h8[1] = arow1[(k0 >> 3) + 2 * half + 1];
    for (int f = 0; f < 4; ++f) {
      const f16* wrow = woh + (size_t)(n0 + f * 16 + col) * Ee + 16 * half;
      prefetch_wgp(wrow + k0 + 64);
      v16h b = *(const v16h*)(wrow + k0);
      acc0[f] = WMMA(a0.v, b, acc0[f]);
      acc1[f] = WMMA(a1.v, b, acc1[f]);
    }
  }
  for (int r = 0; r < 8; ++r)
    for (int f = 0; f < 4; ++f) {
      out[(size_t)(m0 + r + 8 * half) * Ee + n0 + f * 16 + col]      = acc0[f][r];
      out[(size_t)(m0 + 16 + r + 8 * half) * Ee + n0 + f * 16 + col] = acc1[f][r];
    }
}

// ---------------------------------------------------------------------------
extern "C" void kernel_launch(void* const* d_in, const int* in_sizes, int n_in,
                              void* d_out, int out_size, void* d_ws,
                              size_t ws_size, hipStream_t stream) {
  const float* x  = (const float*)d_in[0];
  const float* wq = (const float*)d_in[1];
  const float* wk = (const float*)d_in[2];
  const float* wv = (const float*)d_in[3];
  const float* wo = (const float*)d_in[4];
  float* out = (float*)d_out;

  char* ws = (char*)d_ws;
  f16* xh  = (f16*)(ws + 0);          //  8 MB  [4096,1024]
  f16* wqh = (f16*)(ws + 8388608);    //  2 MB  [1024,1024]
  f16* wkh = (f16*)(ws + 10485760);
  f16* wvh = (f16*)(ws + 12582912);
  f16* woh = (f16*)(ws + 14680064);
  f16* qh  = (f16*)(ws + 16777216);   //  8 MB  [B,H,S,D]
  f16* kh  = (f16*)(ws + 25165824);   //  8 MB  [B,H,S,D]
  f16* vt  = (f16*)(ws + 33554432);   //  8 MB  [B,H,D,S]
  f16* ah  = (f16*)(ws + 41943040);   //  8 MB  [B,S,E]

  // 1) stage to f16
  {
    size_t total = (size_t)Mrows * Ee + 4ull * Ee * Ee;   // 8,388,608
    int blocks = (int)((total + 255) / 256);
    cvt_kernel<<<blocks, 256, 0, stream>>>(x, wq, wk, wv, wo,
                                           xh, wqh, wkh, wvh, woh);
  }
  // 2) QKV + RoPE: 128 m-tiles(32) x 64 n-tiles = 8192 waves, 8 waves/block
  qkv_kernel<<<1024, 256, 0, stream>>>(xh, wqh, wkh, wvh, qh, kh, vt);
  // 3) flash attention: 32 bh x 128 q-tiles = 4096 waves, 4 waves/block
  attn_kernel<<<1024, 128, 0, stream>>>(qh, kh, vt, ah);
  // 4) output projection: 128 m-tiles(32) x 16 n-groups = 2048 waves, 8/block
  proj_kernel<<<256, 256, 0, stream>>>(ah, woh, out);
}
